// PhoenixMambaV2_48679159333166
// MI455X (gfx1250) — compile-verified, matrix operand order checked by reading, not run on hardware
//
#include <hip/hip_runtime.h>
#include <math.h>

typedef _Float16 v16h __attribute__((ext_vector_type(16)));
typedef float    v8f  __attribute__((ext_vector_type(8)));

// ---------------------------------------------------------------------------
// CDNA5 f16 WMMA fragment K-offset (A 16x32 / B 32x16, wave32):
// lanes 0-15 hold K {0..7 | 16..23}, lanes 16-31 hold K {8..15 | 24..31}
// ---------------------------------------------------------------------------
__device__ __forceinline__ int frag_koff(int lane, int v) {
    return ((v < 4) ? 0 : 16) + ((lane >= 16) ? 8 : 0) + 2 * (v & 3);
}
__device__ __forceinline__ float silu_f(float x) { return x / (1.0f + __expf(-x)); }
__device__ __forceinline__ float softplus_f(float x) {
    return (x > 20.0f) ? x : log1pf(__expf(x));
}

// ---------------------------------------------------------------------------
// Pack a row-major f32 K x N weight into zero-padded f16 WMMA B-fragments:
// P[((tn*kSlabs + kt)*32 + lane)*16 + (2v+e)] = W[kt*32+koff(lane,v)+e, tn*16+lane%16]
// ---------------------------------------------------------------------------
__global__ void pm2_pack_w_k(const float* __restrict__ W, _Float16* __restrict__ P,
                             int K, int N, int kSlabs, int total)
{
    int tid = blockIdx.x * blockDim.x + threadIdx.x;
    if (tid >= total) return;
    int j    = tid & 15;
    int lane = (tid >> 4) & 31;
    int slab = tid >> 9;
    int kt   = slab % kSlabs;
    int tn   = slab / kSlabs;
    int v = j >> 1, e = j & 1;
    int k = kt * 32 + frag_koff(lane, v) + e;
    int n = tn * 16 + (lane & 15);
    float val = (k < K && n < N) ? W[(size_t)k * N + n] : 0.0f;
    P[tid] = (_Float16)val;
}

// Per-k conv gather table: {addr delta, dy, dx, 0}; OOB ks get dy=-1e5 (mask off)
__global__ void pm2_conv_tab_k(int4* __restrict__ tab, int Kpad, int K, int Cin, int Wd)
{
    int k = blockIdx.x * blockDim.x + threadIdx.x;
    if (k >= Kpad) return;
    if (k < K) {
        int t = k / Cin, ci = k - t * Cin;
        int ky = t / 3, kx = t - ky * 3;
        tab[k] = make_int4(((ky - 1) * Wd + (kx - 1)) * Cin + ci, ky - 1, kx - 1, 0);
    } else {
        tab[k] = make_int4(0, -100000, -100000, 0);
    }
}

// ---------------------------------------------------------------------------
// WMMA GEMM: C[M,N] = act(A[M,K] @ W + bias) (+residual). W pre-packed f16.
// 16(M) x 64(N) per wave, software-pipelined: slab kt+1 fragments are loaded
// before the 4 WMMAs of slab kt issue, so loads overlap matrix math.
// ---------------------------------------------------------------------------
__global__ __launch_bounds__(256) void pm2_gemm_v3_k(
    const float* __restrict__ A, int lda,
    const _Float16* __restrict__ Wp,
    const float* __restrict__ bias,
    float* __restrict__ C, int ldc,
    const float* __restrict__ residual,
    int M, int N, int kSlabs, int act, int tilesN, int groupsN, int groupsTotal)
{
    int wave = threadIdx.x >> 5, lane = threadIdx.x & 31;
    int g = blockIdx.x * (blockDim.x >> 5) + wave;
    if (g >= groupsTotal) return;                 // wave-uniform
    int tm  = g / groupsN;
    int tn0 = (g - tm * groupsN) * 4;
    int col = lane & 15;

    const float* Ap = A + (size_t)(tm * 16 + col) * lda;
    int t1 = (tn0 + 1 < tilesN) ? tn0 + 1 : tilesN - 1;
    int t2 = (tn0 + 2 < tilesN) ? tn0 + 2 : tilesN - 1;
    int t3 = (tn0 + 3 < tilesN) ? tn0 + 3 : tilesN - 1;
    const v16h* p0 = (const v16h*)Wp + (size_t)tn0 * kSlabs * 32 + lane;
    const v16h* p1 = (const v16h*)Wp + (size_t)t1  * kSlabs * 32 + lane;
    const v16h* p2 = (const v16h*)Wp + (size_t)t2  * kSlabs * 32 + lane;
    const v16h* p3 = (const v16h*)Wp + (size_t)t3  * kSlabs * 32 + lane;

    auto loadA = [&](const float* base) {
        v16h af;
#pragma unroll
        for (int v = 0; v < 8; ++v) {
            float2 t = *(const float2*)(base + frag_koff(lane, v));
            af[2 * v]     = (_Float16)t.x;
            af[2 * v + 1] = (_Float16)t.y;
        }
        return af;
    };

    v8f acc0 = {}, acc1 = {}, acc2 = {}, acc3 = {};
    v16h ca  = loadA(Ap);
    v16h cb0 = *p0, cb1 = *p1, cb2 = *p2, cb3 = *p3;
    for (int kt = 0; kt + 1 < kSlabs; ++kt) {
        Ap += 32; p0 += 32; p1 += 32; p2 += 32; p3 += 32;
        v16h na  = loadA(Ap);                     // prefetch slab kt+1
        v16h nb0 = *p0, nb1 = *p1, nb2 = *p2, nb3 = *p3;
        acc0 = __builtin_amdgcn_wmma_f32_16x16x32_f16(false, ca, false, cb0, (short)0, acc0, false, false);
        acc1 = __builtin_amdgcn_wmma_f32_16x16x32_f16(false, ca, false, cb1, (short)0, acc1, false, false);
        acc2 = __builtin_amdgcn_wmma_f32_16x16x32_f16(false, ca, false, cb2, (short)0, acc2, false, false);
        acc3 = __builtin_amdgcn_wmma_f32_16x16x32_f16(false, ca, false, cb3, (short)0, acc3, false, false);
        ca = na; cb0 = nb0; cb1 = nb1; cb2 = nb2; cb3 = nb3;
    }
    acc0 = __builtin_amdgcn_wmma_f32_16x16x32_f16(false, ca, false, cb0, (short)0, acc0, false, false);
    acc1 = __builtin_amdgcn_wmma_f32_16x16x32_f16(false, ca, false, cb1, (short)0, acc1, false, false);
    acc2 = __builtin_amdgcn_wmma_f32_16x16x32_f16(false, ca, false, cb2, (short)0, acc2, false, false);
    acc3 = __builtin_amdgcn_wmma_f32_16x16x32_f16(false, ca, false, cb3, (short)0, acc3, false, false);

    v8f accs[4] = {acc0, acc1, acc2, acc3};
#pragma unroll
    for (int i = 0; i < 4; ++i) {
        int tn = tn0 + i;
        if (tn >= tilesN) break;
        int n = tn * 16 + col;
        if (n >= N) continue;
        float bv = bias ? bias[n] : 0.0f;
#pragma unroll
        for (int v = 0; v < 8; ++v) {
            int mr = tm * 16 + ((lane < 16) ? 0 : 8) + v;
            if (mr < M) {
                float x = accs[i][v] + bv;
                if (act == 1)      x = silu_f(x);
                else if (act == 2) x = softplus_f(x);
                if (residual) x += residual[(size_t)mr * ldc + n];
                C[(size_t)mr * ldc + n] = x;
            }
        }
    }
}

// ---------------------------------------------------------------------------
// 3x3 SAME conv as implicit GEMM. Branchless halo gather (ktab + clamp +
// cndmask), pre-packed f16 weights, 16x64 tile per wave, software-pipelined.
// ---------------------------------------------------------------------------
__global__ __launch_bounds__(256) void pm2_conv_v3_k(
    const float* __restrict__ x, const _Float16* __restrict__ Wp,
    const float* __restrict__ bias, const int4* __restrict__ tab,
    float* __restrict__ y,
    int H, int Wd, int Cin, int Cout,
    int kSlabs, int tilesN, int groupsN, int groupsTotal)
{
    int wave = threadIdx.x >> 5, lane = threadIdx.x & 31;
    int g = blockIdx.x * (blockDim.x >> 5) + wave;
    if (g >= groupsTotal) return;                 // wave-uniform
    int tm  = g / groupsN;
    int tn0 = (g - tm * groupsN) * 4;
    int col = lane & 15;

    int m   = tm * 16 + col;                      // pixel id (M mult of 16)
    int bi  = m / (H * Wd);
    int rem = m - bi * H * Wd;
    int py  = rem / Wd;
    int px  = rem - py * Wd;
    int pixBase = ((bi * H + py) * Wd + px) * Cin;

    int t1 = (tn0 + 1 < tilesN) ? tn0 + 1 : tilesN - 1;
    int t2 = (tn0 + 2 < tilesN) ? tn0 + 2 : tilesN - 1;
    int t3 = (tn0 + 3 < tilesN) ? tn0 + 3 : tilesN - 1;
    const v16h* p0 = (const v16h*)Wp + (size_t)tn0 * kSlabs * 32 + lane;
    const v16h* p1 = (const v16h*)Wp + (size_t)t1  * kSlabs * 32 + lane;
    const v16h* p2 = (const v16h*)Wp + (size_t)t2  * kSlabs * 32 + lane;
    const v16h* p3 = (const v16h*)Wp + (size_t)t3  * kSlabs * 32 + lane;

    auto gatherA = [&](int kbase) {
        v16h af;
#pragma unroll
        for (int v = 0; v < 8; ++v) {
            int k0 = kbase + frag_koff(lane, v);
#pragma unroll
            for (int e = 0; e < 2; ++e) {
                int4 en = tab[k0 + e];
                int iy = py + en.y, ix = px + en.z;
                bool in = ((unsigned)iy < (unsigned)H) && ((unsigned)ix < (unsigned)Wd);
                float raw = x[in ? (pixBase + en.x) : 0];   // unconditional load
                af[2 * v + e] = (_Float16)(in ? raw : 0.0f);
            }
        }
        return af;
    };

    v8f acc0 = {}, acc1 = {}, acc2 = {}, acc3 = {};
    v16h ca  = gatherA(0);
    v16h cb0 = *p0, cb1 = *p1, cb2 = *p2, cb3 = *p3;
    for (int kt = 0; kt + 1 < kSlabs; ++kt) {
        p0 += 32; p1 += 32; p2 += 32; p3 += 32;
        v16h na  = gatherA((kt + 1) * 32);        // prefetch slab kt+1
        v16h nb0 = *p0, nb1 = *p1, nb2 = *p2, nb3 = *p3;
        acc0 = __builtin_amdgcn_wmma_f32_16x16x32_f16(false, ca, false, cb0, (short)0, acc0, false, false);
        acc1 = __builtin_amdgcn_wmma_f32_16x16x32_f16(false, ca, false, cb1, (short)0, acc1, false, false);
        acc2 = __builtin_amdgcn_wmma_f32_16x16x32_f16(false, ca, false, cb2, (short)0, acc2, false, false);
        acc3 = __builtin_amdgcn_wmma_f32_16x16x32_f16(false, ca, false, cb3, (short)0, acc3, false, false);
        ca = na; cb0 = nb0; cb1 = nb1; cb2 = nb2; cb3 = nb3;
    }
    acc0 = __builtin_amdgcn_wmma_f32_16x16x32_f16(false, ca, false, cb0, (short)0, acc0, false, false);
    acc1 = __builtin_amdgcn_wmma_f32_16x16x32_f16(false, ca, false, cb1, (short)0, acc1, false, false);
    acc2 = __builtin_amdgcn_wmma_f32_16x16x32_f16(false, ca, false, cb2, (short)0, acc2, false, false);
    acc3 = __builtin_amdgcn_wmma_f32_16x16x32_f16(false, ca, false, cb3, (short)0, acc3, false, false);

    v8f accs[4] = {acc0, acc1, acc2, acc3};
#pragma unroll
    for (int i = 0; i < 4; ++i) {
        int tn = tn0 + i;
        if (tn >= tilesN) break;
        int n = tn * 16 + col;                    // Cout mult of 16: in-bounds
        float bv = bias[n];
#pragma unroll
        for (int v = 0; v < 8; ++v) {
            int mr = tm * 16 + ((lane < 16) ? 0 : 8) + v;
            y[(size_t)mr * Cout + n] = accs[i][v] + bv;
        }
    }
}

// ---------------------------------------------------------------------------
// Wave-per-row LayerNorm
// ---------------------------------------------------------------------------
__global__ __launch_bounds__(256) void pm2_layernorm_k(
    const float* __restrict__ x, const float* __restrict__ g,
    const float* __restrict__ b, float* __restrict__ y,
    int rows, int C, float eps)
{
    int wave = threadIdx.x >> 5, lane = threadIdx.x & 31;
    int row = blockIdx.x * (blockDim.x >> 5) + wave;
    if (row >= rows) return;
    const float* xr = x + (size_t)row * C;
    float s = 0.0f;
    for (int i = lane; i < C; i += 32) s += xr[i];
    for (int msk = 16; msk >= 1; msk >>= 1) s += __shfl_xor(s, msk, 32);
    float mean = s / (float)C;
    float v = 0.0f;
    for (int i = lane; i < C; i += 32) { float d = xr[i] - mean; v += d * d; }
    for (int msk = 16; msk >= 1; msk >>= 1) v += __shfl_xor(v, msk, 32);
    float rstd = rsqrtf(v / (float)C + eps);
    float* yr = y + (size_t)row * C;
    for (int i = lane; i < C; i += 32)
        yr[i] = (xr[i] - mean) * rstd * g[i] + b[i];
}

// ---------------------------------------------------------------------------
// Selective scan: thread per (b,d,n); 16 state dims in 16 adjacent lanes.
// ---------------------------------------------------------------------------
__global__ __launch_bounds__(256) void pm2_scan_k(
    const float* __restrict__ uz, int ldu,
    const float* __restrict__ dt,
    const float* __restrict__ proj, int ldp, int r,
    const float* __restrict__ A_log, const float* __restrict__ Dp,
    float* __restrict__ y, int Bn, int L, int di)
{
    int tid = blockIdx.x * blockDim.x + threadIdx.x;
    if (tid >= Bn * di * 16) return;
    int n  = tid & 15;
    int dd = (tid >> 4) % di;
    int b  = tid / (di * 16);

    float a  = -__expf(A_log[dd * 16 + n]);
    float Dv = Dp[dd];
    float h  = 0.0f;
    const float* pdt = dt   + (size_t)b * L * di  + dd;
    const float* pu  = uz   + (size_t)b * L * ldu + dd;
    const float* pB  = proj + (size_t)b * L * ldp + r + n;
    float*       py  = y    + (size_t)b * L * di  + dd;
    for (int l = 0; l < L; ++l) {
        float dtv = *pdt;
        float uv  = *pu;
        float Bv  = pB[0];
        float Cv  = pB[16];
        h = __expf(dtv * a) * h + (dtv * uv) * Bv;
        float contrib = h * Cv;
        for (int msk = 8; msk >= 1; msk >>= 1)
            contrib += __shfl_xor(contrib, msk, 32);
        if (n == 0) *py = contrib + uv * Dv;
        pdt += di; pu += ldu; pB += ldp; py += di;
    }
}

// ---------------------------------------------------------------------------
// Elementwise helpers
// ---------------------------------------------------------------------------
__global__ void pm2_silu_cols_k(float* __restrict__ uz, int total, int di) {
    int i = blockIdx.x * blockDim.x + threadIdx.x;
    if (i >= total) return;
    int row = i / di, c = i - row * di;
    float* p = uz + (size_t)row * (2 * di) + c;
    *p = silu_f(*p);
}

__global__ void pm2_gate_mul_k(float* __restrict__ y, const float* __restrict__ uz,
                               int total, int di) {
    int i = blockIdx.x * blockDim.x + threadIdx.x;
    if (i >= total) return;
    int row = i / di, c = i - row * di;
    y[i] *= silu_f(uz[(size_t)row * (2 * di) + di + c]);
}

__global__ void pm2_maxpool2_k(const float* __restrict__ x, float* __restrict__ y,
                               int Bn, int H, int Wd, int C) {
    int Ho = H >> 1, Wo = Wd >> 1;
    int total = Bn * Ho * Wo * C;
    int i = blockIdx.x * blockDim.x + threadIdx.x;
    if (i >= total) return;
    int c = i % C; int t = i / C;
    int ox = t % Wo; t /= Wo;
    int oy = t % Ho; int b = t / Ho;
    size_t base = (size_t)((b * H + 2 * oy) * Wd + 2 * ox) * C + c;
    float m0 = fmaxf(x[base], x[base + C]);
    float m1 = fmaxf(x[base + (size_t)Wd * C], x[base + (size_t)Wd * C + C]);
    y[i] = fmaxf(m0, m1);
}

// ---------------------------------------------------------------------------
// Attention pieces (L=256, d=512, 8 heads x 64)
// ---------------------------------------------------------------------------
__global__ void pm2_scores_k(const float* __restrict__ q, const float* __restrict__ k,
                             float* __restrict__ s) {
    int tid = blockIdx.x * blockDim.x + threadIdx.x;      // 2*8*256*256 = 2^20
    if (tid >= (1 << 20)) return;
    int kk = tid & 255, qq = (tid >> 8) & 255, h = (tid >> 16) & 7, b = tid >> 19;
    const float* qp = q + ((size_t)(b * 256 + qq) * 512) + h * 64;
    const float* kp = k + ((size_t)(b * 256 + kk) * 512) + h * 64;
    float acc = 0.0f;
    for (int i = 0; i < 64; ++i) acc += qp[i] * kp[i];
    s[tid] = acc * 0.125f;
}

__global__ __launch_bounds__(256) void pm2_softmax_rows_k(float* __restrict__ s,
                                                          int rows, int C) {
    int wave = threadIdx.x >> 5, lane = threadIdx.x & 31;
    int row = blockIdx.x * (blockDim.x >> 5) + wave;
    if (row >= rows) return;
    float* r = s + (size_t)row * C;
    float mx = -3.4e38f;
    for (int i = lane; i < C; i += 32) mx = fmaxf(mx, r[i]);
    for (int msk = 16; msk >= 1; msk >>= 1) mx = fmaxf(mx, __shfl_xor(mx, msk, 32));
    float sum = 0.0f;
    for (int i = lane; i < C; i += 32) { float e = __expf(r[i] - mx); r[i] = e; sum += e; }
    for (int msk = 16; msk >= 1; msk >>= 1) sum += __shfl_xor(sum, msk, 32);
    float inv = 1.0f / sum;
    for (int i = lane; i < C; i += 32) r[i] *= inv;
}

__global__ void pm2_attn_av_k(const float* __restrict__ s, const float* __restrict__ v,
                              float* __restrict__ o) {
    int tid = blockIdx.x * blockDim.x + threadIdx.x;      // 2*256*8*64 = 2^18
    if (tid >= (1 << 18)) return;
    int i = tid & 63, h = (tid >> 6) & 7, q = (tid >> 9) & 255, b = tid >> 17;
    const float* sp = s + (size_t)((b * 8 + h) * 256 + q) * 256;
    const float* vp = v + (size_t)(b * 256) * 512 + h * 64 + i;
    float acc = 0.0f;
    for (int kk = 0; kk < 256; ++kk) acc += sp[kk] * vp[(size_t)kk * 512];
    o[(size_t)(b * 256 + q) * 512 + h * 64 + i] = acc;
}

__global__ void pm2_meanpool_k(const float* __restrict__ xf, const float* __restrict__ xa,
                               float* __restrict__ xp) {
    int tid = blockIdx.x * blockDim.x + threadIdx.x;      // 2*512
    if (tid >= 1024) return;
    int b = tid >> 9, c = tid & 511;
    float s = 0.0f;
    for (int l = 0; l < 256; ++l) {
        size_t idx = (size_t)(b * 256 + l) * 512 + c;
        s += xf[idx] + xa[idx];
    }
    xp[(size_t)b * 512 + c] = s * (1.0f / 256.0f);
}

__global__ void pm2_softmax_out_k(const float* __restrict__ lg, float* __restrict__ out) {
    int row = blockIdx.x * blockDim.x + threadIdx.x;
    if (row >= 2) return;
    float mx = -3.4e38f;
    for (int i = 0; i < 10; ++i) mx = fmaxf(mx, lg[row * 10 + i]);
    float e[10], s = 0.0f;
    for (int i = 0; i < 10; ++i) { e[i] = __expf(lg[row * 10 + i] - mx); s += e[i]; }
    float inv = 1.0f / s;
    for (int i = 0; i < 10; ++i) out[row * 10 + i] = e[i] * inv;
}

// ---------------------------------------------------------------------------
// Host orchestration
// ---------------------------------------------------------------------------
extern "C" void kernel_launch(void* const* d_in, const int* in_sizes, int n_in,
                              void* d_out, int out_size, void* d_ws, size_t ws_size,
                              hipStream_t stream)
{
    (void)in_sizes; (void)n_in; (void)out_size; (void)ws_size;
    const int Bn = 2;

    // ---- float workspace ------------------------------------------------
    float* ws = (float*)d_ws;
    size_t off = 0;
    auto carve = [&](size_t n) { float* p = ws + off; off += n; return p; };
    float* X0  = carve(2097152);
    float* C1  = carve(2097152);
    float* LNb = carve(2097152);
    float* UZ  = carve(8388608);
    float* PR  = carve(1310720);
    float* DT  = carve(4194304);
    float* Yb  = carve(4194304);
    float* MO  = carve(2097152);
    float* Sc  = carve(1048576);
    float* Qb  = carve(262144);
    float* Kb  = carve(262144);
    float* Vb  = carve(262144);
    float* OP  = carve(262144);
    float* XA  = carve(262144);
    float* XP  = carve(8192);      // 16 x 512, rows 2..15 zeroed
    float* LG  = carve(64);

    // ---- f16 packed-weight pool -----------------------------------------
    _Float16* hp = (_Float16*)(ws + off);
    size_t hoff = 0;
    auto packW = [&](const float* W, int K, int N) {
        int kSlabs = (K + 31) / 32, tilesN = (N + 15) / 16;
        int total = tilesN * kSlabs * 512;
        _Float16* P = hp + hoff; hoff += (size_t)total;
        pm2_pack_w_k<<<(total + 255) / 256, 256, 0, stream>>>(W, P, K, N, kSlabs, total);
        return P;
    };

    const size_t H16_BUDGET = 6u * 1024 * 1024;          // 6M f16 = 12MB > ~4.9M used
    int4* tabPool = (int4*)(hp + H16_BUDGET);
    size_t toff = 0;
    auto makeTab = [&](int K, int Cin, int Wd) {
        int Kpad = ((K + 31) / 32) * 32;
        int4* t = tabPool + toff; toff += (size_t)Kpad;
        pm2_conv_tab_k<<<(Kpad + 255) / 256, 256, 0, stream>>>(t, Kpad, K, Cin, Wd);
        return t;
    };

    auto gemm = [&](const float* A, int lda, const _Float16* Wp, const float* bias,
                    float* C, int ldc, const float* res,
                    int M, int N, int K, int act) {
        int tilesM = (M + 15) / 16, tilesN = (N + 15) / 16, kSlabs = (K + 31) / 32;
        int groupsN = (tilesN + 3) / 4;
        int groups = tilesM * groupsN;
        pm2_gemm_v3_k<<<(groups + 7) / 8, 256, 0, stream>>>(
            A, lda, Wp, bias, C, ldc, res, M, N, kSlabs, act, tilesN, groupsN, groups);
    };
    auto conv = [&](const float* x, const _Float16* Wp, const float* b, const int4* tab,
                    float* y, int H, int Wd, int Cin, int Cout) {
        int M = Bn * H * Wd, K = 9 * Cin;
        int tilesN = Cout / 16, kSlabs = (K + 31) / 32;
        int groupsN = (tilesN + 3) / 4;
        int groups = (M / 16) * groupsN;
        pm2_conv_v3_k<<<(groups + 7) / 8, 256, 0, stream>>>(
            x, Wp, b, tab, y, H, Wd, Cin, Cout, kSlabs, tilesN, groupsN, groups);
    };
    auto ew = [&](int total) { return (total + 255) / 256; };

    // ---- stage constants -------------------------------------------------
    const int stH[4] = {128, 64, 32, 16};
    const int stCin[4] = {64, 64, 128, 256};
    const int stD[4] = {64, 128, 256, 512};

    // ---- pack all weights (on-stream, before first use) ------------------
    const _Float16* aggP = packW((const float*)d_in[1], 27, 64);
    const int4*     aggT = makeTab(27, 3, 128);
    const _Float16 *cP[4]; const int4* cT[4];
    const _Float16 *WiP[4], *WxP[4], *WdtP[4], *WoP[4];
    for (int s = 0; s < 4; ++s) {
        int d = stD[s], cin = stCin[s], di = 2 * d, r = (d + 15) / 16;
        int base = 3 + s * 13;
        cP[s]   = packW((const float*)d_in[base + 0], 9 * cin, d);
        cT[s]   = makeTab(9 * cin, cin, stH[s]);
        WiP[s]  = packW((const float*)d_in[base + 4], d, 2 * di);
        WxP[s]  = packW((const float*)d_in[base + 6], di, r + 32);
        WdtP[s] = packW((const float*)d_in[base + 7], r, di);
        WoP[s]  = packW((const float*)d_in[base + 11], di, d);
    }
    const _Float16* WqP  = packW((const float*)d_in[57], 512, 512);
    const _Float16* WkP  = packW((const float*)d_in[59], 512, 512);
    const _Float16* WvP  = packW((const float*)d_in[61], 512, 512);
    const _Float16* WaoP = packW((const float*)d_in[63], 512, 512);
    const _Float16* WcP  = packW((const float*)d_in[65], 512, 10);

    hipMemsetAsync(XP, 0, 8192 * sizeof(float), stream);  // zero cls input padding

    // ---- aggregation conv: 3 -> 64 at 128x128 ---------------------------
    conv((const float*)d_in[0], aggP, (const float*)d_in[2], aggT, X0, 128, 128, 3, 64);

    // ---- stages ----------------------------------------------------------
    for (int s = 0; s < 4; ++s) {
        int H = stH[s], Wd = stH[s], d = stD[s], cin = stCin[s];
        int L = H * Wd, BL = Bn * L;
        int di = 2 * d, r = (d + 15) / 16, ldp = r + 32;
        int base = 3 + s * 13;
        const float* cb   = (const float*)d_in[base + 1];
        const float* lng  = (const float*)d_in[base + 2];
        const float* lnb  = (const float*)d_in[base + 3];
        const float* bi   = (const float*)d_in[base + 5];
        const float* bdt  = (const float*)d_in[base + 8];
        const float* Alog = (const float*)d_in[base + 9];
        const float* Dp   = (const float*)d_in[base + 10];
        const float* bo   = (const float*)d_in[base + 12];

        conv(X0, cP[s], cb, cT[s], C1, H, Wd, cin, d);                      // x
        pm2_layernorm_k<<<(BL + 7) / 8, 256, 0, stream>>>(C1, lng, lnb, LNb,
                                                          BL, d, 1e-5f);    // h
        gemm(LNb, d, WiP[s], bi, UZ, 2 * di, nullptr, BL, 2 * di, d, 0);    // u|z
        pm2_silu_cols_k<<<ew(BL * di), 256, 0, stream>>>(UZ, BL * di, di);  // silu(u)
        gemm(UZ, 2 * di, WxP[s], nullptr, PR, ldp, nullptr, BL, ldp, di, 0);// proj
        gemm(PR, ldp, WdtP[s], bdt, DT, di, nullptr, BL, di, r, 2);         // softplus dt
        {
            int total = Bn * di * 16;
            pm2_scan_k<<<(total + 255) / 256, 256, 0, stream>>>(
                UZ, 2 * di, DT, PR, ldp, r, Alog, Dp, Yb, Bn, L, di);       // y
        }
        pm2_gate_mul_k<<<ew(BL * di), 256, 0, stream>>>(Yb, UZ, BL * di, di);
        gemm(Yb, di, WoP[s], bo, MO, d, C1, BL, d, di, 0);                  // x + y@Wo+bo
        if (s < 3) {
            int total = Bn * (H / 2) * (Wd / 2) * d;
            pm2_maxpool2_k<<<ew(total), 256, 0, stream>>>(MO, X0, Bn, H, Wd, d);
        }
    }

    // ---- attention tail (xf = MO: [2,256,512]) ---------------------------
    const float* ag  = (const float*)d_in[55];
    const float* ab  = (const float*)d_in[56];
    const float* bq  = (const float*)d_in[58];
    const float* bk  = (const float*)d_in[60];
    const float* bv  = (const float*)d_in[62];
    const float* bao = (const float*)d_in[64];
    const float* bc  = (const float*)d_in[66];

    pm2_layernorm_k<<<(512 + 7) / 8, 256, 0, stream>>>(MO, ag, ab, LNb, 512, 512, 1e-3f);
    gemm(LNb, 512, WqP, bq, Qb, 512, nullptr, 512, 512, 512, 0);
    gemm(LNb, 512, WkP, bk, Kb, 512, nullptr, 512, 512, 512, 0);
    gemm(LNb, 512, WvP, bv, Vb, 512, nullptr, 512, 512, 512, 0);
    pm2_scores_k<<<(1 << 20) / 256, 256, 0, stream>>>(Qb, Kb, Sc);
    pm2_softmax_rows_k<<<(4096 + 7) / 8, 256, 0, stream>>>(Sc, 4096, 256);
    pm2_attn_av_k<<<(1 << 18) / 256, 256, 0, stream>>>(Sc, Vb, OP);
    gemm(OP, 512, WaoP, bao, XA, 512, nullptr, 512, 512, 512, 0);
    pm2_meanpool_k<<<4, 256, 0, stream>>>(MO, XA, XP);
    gemm(XP, 512, WcP, bc, LG, 10, nullptr, 2, 10, 512, 0);
    pm2_softmax_out_k<<<1, 64, 0, stream>>>(LG, (float*)d_out);
}
